// GAT_GraphSAGE_12781822673344
// MI455X (gfx1250) — matrix operand order repaired and checked by reading, not compiled, not run on hardware
//
#include <hip/hip_runtime.h>
#include <hip/hip_bf16.h>

// Problem constants (from reference)
#define N_NODES  10000
#define NPAD     10016      // N padded to multiple of 32 for key/value chunks
#define E_EDGES  320000
#define G_GRAPHS 256
#define F_DIM    35
#define FPAD     64         // feature dim padded for K=32 WMMA steps
#define H1_DIM   1500
#define H2_DIM   128

typedef __attribute__((ext_vector_type(16))) _Float16 v16h;
typedef __attribute__((ext_vector_type(8)))  float    v8f;
typedef _Float16 half_t;

// ---------------------------------------------------------------------------
// Zero scratch accumulators (agg, deg, gmax, gsum, gcnt are contiguous)
// ---------------------------------------------------------------------------
__global__ void zero_f32_kernel(float* __restrict__ p, int n) {
  for (int i = blockIdx.x * blockDim.x + threadIdx.x; i < n;
       i += gridDim.x * blockDim.x)
    p[i] = 0.0f;
}

// ---------------------------------------------------------------------------
// Q,K,V projections.  Writes Qh [NPAD][FPAD] f16 (zero padded),
// Kf [N][F] f32, Vt [FPAD][NPAD] f16 (transposed, zero padded).
// ---------------------------------------------------------------------------
__global__ void qkv_kernel(const float* __restrict__ x,
                           const float* __restrict__ Wq, const float* __restrict__ bq,
                           const float* __restrict__ Wk, const float* __restrict__ bk,
                           const float* __restrict__ Wv, const float* __restrict__ bv,
                           half_t* __restrict__ Qh, float* __restrict__ Kf,
                           half_t* __restrict__ Vt) {
  int idx = blockIdx.x * blockDim.x + threadIdx.x;
  int n = idx / FPAD, o = idx % FPAD;
  if (n >= NPAD) return;
  float q = 0.f, k = 0.f, v = 0.f;
  if (n < N_NODES && o < F_DIM) {
    const float* xr = x + n * F_DIM;
    #pragma unroll 5
    for (int i = 0; i < F_DIM; ++i) {
      float xi = xr[i];
      q += xi * Wq[i * F_DIM + o];
      k += xi * Wk[i * F_DIM + o];
      v += xi * Wv[i * F_DIM + o];
    }
    q += bq[o]; k += bk[o]; v += bv[o];
    Kf[n * F_DIM + o] = k;
  }
  Qh[(size_t)n * FPAD + o] = (half_t)q;
  Vt[(size_t)o * NPAD + n] = (half_t)v;
}

// ---------------------------------------------------------------------------
// Center-tap "Conv1d" layers: Kc3 = K @ Wc3[:,:,1].T + bc3 ; Kc5 similarly.
// Wc3 flat layout: (out, in, 3) -> [o*F*3 + i*3 + tap]
// ---------------------------------------------------------------------------
__global__ void conv_kernel(const float* __restrict__ Kf,
                            const float* __restrict__ Wc3, const float* __restrict__ bc3,
                            const float* __restrict__ Wc5, const float* __restrict__ bc5,
                            float* __restrict__ Kc3, float* __restrict__ Kc5) {
  int idx = blockIdx.x * blockDim.x + threadIdx.x;
  if (idx >= N_NODES * F_DIM) return;
  int n = idx / F_DIM, o = idx % F_DIM;
  const float* kr = Kf + n * F_DIM;
  float a3 = bc3[o], a5 = bc5[o];
  #pragma unroll 5
  for (int i = 0; i < F_DIM; ++i) {
    float ki = kr[i];
    a3 += ki * Wc3[(o * F_DIM + i) * 3 + 1];
    a5 += ki * Wc5[(o * F_DIM + i) * 5 + 2];
  }
  Kc3[idx] = a3;
  Kc5[idx] = a5;
}

// ---------------------------------------------------------------------------
// K_new = [Kc3, Kc5, K] @ Wlt + blt   ->  Knh [NPAD][FPAD] f16 (zero padded)
// ---------------------------------------------------------------------------
__global__ void knew_kernel(const float* __restrict__ Kc3, const float* __restrict__ Kc5,
                            const float* __restrict__ Kf,
                            const float* __restrict__ Wlt, const float* __restrict__ blt,
                            half_t* __restrict__ Knh) {
  int idx = blockIdx.x * blockDim.x + threadIdx.x;
  int n = idx / FPAD, o = idx % FPAD;
  if (n >= NPAD) return;
  float a = 0.f;
  if (n < N_NODES && o < F_DIM) {
    a = blt[o];
    #pragma unroll 5
    for (int j = 0; j < F_DIM; ++j) {
      a += Kc3[n * F_DIM + j] * Wlt[j * F_DIM + o];
      a += Kc5[n * F_DIM + j] * Wlt[(F_DIM + j) * F_DIM + o];
      a += Kf [n * F_DIM + j] * Wlt[(2 * F_DIM + j) * F_DIM + o];
    }
  }
  Knh[(size_t)n * FPAD + o] = (half_t)a;
}

// ---------------------------------------------------------------------------
// Fused dense attention (flash-attention style, never materializes NxN):
//   h = relu(softmax(Q K_new^T / sqrt(F)) @ V)
// One wave (32 lanes) per 16-row Q tile.  WMMA f32 <- f16 16x16x32.
//
// C/D layout (f32 16x16): lane l, vgpr r holds element (M = r + 8*(l>>4),
// N = l&15).  A as loaded here (f16 16x32): lane l row M = l&15, elems t are
// K = 16*(l>>4)+t (contiguous, not AMD's table order).  B as loaded
// (f16 32x16): lane l col N = l&15, elems t hold K = 16*(l>>4)+t.
// ---------------------------------------------------------------------------
__global__ __launch_bounds__(32)
void attn_kernel(const half_t* __restrict__ Qh, const half_t* __restrict__ Knh,
                 const half_t* __restrict__ Vt, float* __restrict__ hout) {
  __shared__ half_t stag[16 * 32];       // P staging: C/D layout -> A layout

  const int lane    = threadIdx.x & 31;
  const int half_id = lane >> 4;
  const int nsub    = lane & 15;
  const int row_base = blockIdx.x * 16;
  const float scale = 0.16903085094570331f;   // 1/sqrt(35)

  // Preload the two K-step A fragments of the Q tile (row-major [node][FPAD])
  v16h aq0, aq1;
  {
    const half_t* qrow = Qh + (size_t)(row_base + nsub) * FPAD;
    const int kb = half_id * 16;
    #pragma unroll
    for (int t = 0; t < 8; ++t) {
      aq0[t]     = qrow[kb + t];
      aq0[8 + t] = qrow[kb + 8 + t];
      aq1[t]     = qrow[32 + kb + t];
      aq1[8 + t] = qrow[32 + kb + 8 + t];
    }
  }

  float m_run[8], s_run[8];
  v8f Oc[4];
  #pragma unroll
  for (int r = 0; r < 8; ++r) { m_run[r] = -3e38f; s_run[r] = 0.f; }
  #pragma unroll
  for (int ft = 0; ft < 4; ++ft)
    #pragma unroll
    for (int e = 0; e < 8; ++e) Oc[ft][e] = 0.f;

  const int nchunks = NPAD / 32;         // 313 chunks of 32 key columns
  for (int ch = 0; ch < nchunks; ++ch) {
    const int cb = ch * 32;

    // Prefetch next chunk's K_new / V tiles into cache (global_prefetch_b8)
    if (ch + 1 < nchunks) {
      __builtin_prefetch(Knh + (size_t)(cb + 32 + nsub) * FPAD, 0, 1);
      __builtin_prefetch(Vt + (size_t)nsub * NPAD + cb + 32, 0, 1);
    }

    float sc[2][8];
    float tmax[8];
    #pragma unroll
    for (int r = 0; r < 8; ++r) tmax[r] = -3e38f;

    // ---- scores: two 16x16 tiles, each = 2 WMMAs over padded K=64 ----
    #pragma unroll
    for (int ct = 0; ct < 2; ++ct) {
      const int col = cb + ct * 16 + nsub;
      const half_t* krow = Knh + (size_t)col * FPAD;
      const int kblock = half_id * 16;
      v16h b0, b1;
      #pragma unroll
      for (int t = 0; t < 16; ++t) {
        b0[t] = krow[kblock + t];
        b1[t] = krow[32 + kblock + t];
      }
      v8f acc;
      #pragma unroll
      for (int e = 0; e < 8; ++e) acc[e] = 0.f;
      acc = __builtin_amdgcn_wmma_f32_16x16x32_f16(false, aq0, false, b0,
                                                   (short)0, acc, false, false);
      acc = __builtin_amdgcn_wmma_f32_16x16x32_f16(false, aq1, false, b1,
                                                   (short)0, acc, false, false);
      const bool oob = (col >= N_NODES);
      #pragma unroll
      for (int r = 0; r < 8; ++r) {
        float s = oob ? -3e38f : acc[r] * scale;
        sc[ct][r] = s;
        tmax[r] = fmaxf(tmax[r], s);
      }
    }

    // ---- row max across the 16 lanes of each half-wave ----
    #pragma unroll
    for (int r = 0; r < 8; ++r) {
      float v = tmax[r];
      v = fmaxf(v, __shfl_xor(v, 1, 32));
      v = fmaxf(v, __shfl_xor(v, 2, 32));
      v = fmaxf(v, __shfl_xor(v, 4, 32));
      v = fmaxf(v, __shfl_xor(v, 8, 32));
      tmax[r] = v;
    }

    float p0[8], p1[8], alpha[8], rsum[8];
    #pragma unroll
    for (int r = 0; r < 8; ++r) {
      float mn = fmaxf(m_run[r], tmax[r]);
      alpha[r] = __expf(m_run[r] - mn);
      m_run[r] = mn;
      p0[r] = __expf(sc[0][r] - mn);
      p1[r] = __expf(sc[1][r] - mn);
      rsum[r] = p0[r] + p1[r];
    }
    #pragma unroll
    for (int r = 0; r < 8; ++r) {
      float v = rsum[r];
      v += __shfl_xor(v, 1, 32);
      v += __shfl_xor(v, 2, 32);
      v += __shfl_xor(v, 4, 32);
      v += __shfl_xor(v, 8, 32);
      s_run[r] = s_run[r] * alpha[r] + v;
    }
    #pragma unroll
    for (int ft = 0; ft < 4; ++ft)
      #pragma unroll
      for (int e = 0; e < 8; ++e) Oc[ft][e] *= alpha[e];

    // ---- bounce P through LDS: C/D layout -> A layout ----
    __syncthreads();                      // WAR vs previous iteration's reads
    #pragma unroll
    for (int r = 0; r < 8; ++r) {
      stag[(r + 8 * half_id) * 32 + nsub]      = (half_t)p0[r];
      stag[(r + 8 * half_id) * 32 + 16 + nsub] = (half_t)p1[r];
    }
    __syncthreads();

    v16h a2;
    {
      const int kb2 = half_id * 16;
      #pragma unroll
      for (int t = 0; t < 8; ++t) {
        a2[t]     = stag[nsub * 32 + kb2 + t];
        a2[8 + t] = stag[nsub * 32 + kb2 + 8 + t];
      }
    }

    // ---- O += P @ V over 4 feature tiles (V stored transposed) ----
    #pragma unroll
    for (int ft = 0; ft < 4; ++ft) {
      const int feat = ft * 16 + nsub;
      const half_t* vrow = Vt + (size_t)feat * NPAD + cb;
      const int kblock = half_id * 16;
      v16h b2;
      #pragma unroll
      for (int t = 0; t < 16; ++t) b2[t] = vrow[kblock + t];
      Oc[ft] = __builtin_amdgcn_wmma_f32_16x16x32_f16(false, a2, false, b2,
                                                      (short)0, Oc[ft], false, false);
    }
  }

  // ---- normalize, relu, write h [N][F] f32 ----
  #pragma unroll
  for (int ft = 0; ft < 4; ++ft) {
    const int feat = ft * 16 + nsub;
    if (feat < F_DIM) {
      #pragma unroll
      for (int r = 0; r < 8; ++r) {
        const int row = row_base + r + 8 * half_id;
        float v = Oc[ft][r] / s_run[r];
        hout[row * F_DIM + feat] = fmaxf(v, 0.f);
      }
    }
  }
}

// ---------------------------------------------------------------------------
// SAGE mean aggregation over incoming edges (atomics)
// ---------------------------------------------------------------------------
__global__ void sage_agg_kernel(const int* __restrict__ ei,
                                const float* __restrict__ h,
                                float* __restrict__ agg, float* __restrict__ deg) {
  int e = blockIdx.x * blockDim.x + threadIdx.x;
  if (e >= E_EDGES) return;
  int src = ei[e];
  int dst = ei[E_EDGES + e];
  atomicAdd(&deg[dst], 1.0f);
  const float* hs = h + src * F_DIM;
  float* ad = agg + dst * F_DIM;
  #pragma unroll 5
  for (int f = 0; f < F_DIM; ++f) atomicAdd(&ad[f], hs[f]);
}

// h2 = relu((agg/deg) @ Wl + bl + h @ Wr)
__global__ void sage_update_kernel(const float* __restrict__ agg,
                                   const float* __restrict__ deg,
                                   const float* __restrict__ h,
                                   const float* __restrict__ Wl, const float* __restrict__ bl,
                                   const float* __restrict__ Wr,
                                   float* __restrict__ h2) {
  int idx = blockIdx.x * blockDim.x + threadIdx.x;
  if (idx >= N_NODES * F_DIM) return;
  int n = idx / F_DIM, o = idx % F_DIM;
  float d = fmaxf(deg[n], 1.0f);
  float inv_d = 1.0f / d;
  float acc = bl[o];
  #pragma unroll 5
  for (int i = 0; i < F_DIM; ++i) {
    acc += (agg[n * F_DIM + i] * inv_d) * Wl[i * F_DIM + o];
    acc += h[n * F_DIM + i] * Wr[i * F_DIM + o];
  }
  h2[idx] = fmaxf(acc, 0.0f);
}

// ---------------------------------------------------------------------------
// Per-graph max + mean pooling (h2 >= 0, so float-as-int atomicMax is valid)
// ---------------------------------------------------------------------------
__global__ void pool_kernel(const int* __restrict__ batch,
                            const float* __restrict__ h2,
                            int* __restrict__ gmaxi, float* __restrict__ gsum,
                            float* __restrict__ gcnt) {
  int idx = blockIdx.x * blockDim.x + threadIdx.x;
  if (idx >= N_NODES * F_DIM) return;
  int n = idx / F_DIM, o = idx % F_DIM;
  int g = batch[n];
  float v = h2[idx];
  atomicMax(&gmaxi[g * F_DIM + o], __float_as_int(v));
  atomicAdd(&gsum[g * F_DIM + o], v);
  if (o == 0) atomicAdd(&gcnt[g], 1.0f);
}

__global__ void gvec_kernel(const int* __restrict__ gmaxi,
                            const float* __restrict__ gsum,
                            const float* __restrict__ gcnt,
                            float* __restrict__ gvec) {
  int idx = blockIdx.x * blockDim.x + threadIdx.x;
  if (idx >= G_GRAPHS * 2 * F_DIM) return;
  int g = idx / (2 * F_DIM), k = idx % (2 * F_DIM);
  float v;
  if (k < F_DIM) {
    v = __int_as_float(gmaxi[g * F_DIM + k]);     // init 0 handles empty graphs
  } else {
    float c = fmaxf(gcnt[g], 1.0f);
    v = gsum[g * F_DIM + (k - F_DIM)] / c;
  }
  gvec[idx] = v;
}

// ---------------------------------------------------------------------------
// Head MLP
// ---------------------------------------------------------------------------
__global__ void mlp1_kernel(const float* __restrict__ gvec,
                            const float* __restrict__ W1, const float* __restrict__ b1,
                            float* __restrict__ hid1) {
  int idx = blockIdx.x * blockDim.x + threadIdx.x;
  if (idx >= G_GRAPHS * H1_DIM) return;
  int g = idx / H1_DIM, j = idx % H1_DIM;
  float acc = b1[j];
  const float* gv = gvec + g * 2 * F_DIM;
  #pragma unroll 10
  for (int k = 0; k < 2 * F_DIM; ++k) acc += gv[k] * W1[k * H1_DIM + j];
  hid1[idx] = fmaxf(acc, 0.0f);
}

__global__ void mlp2_kernel(const float* __restrict__ hid1,
                            const float* __restrict__ W2, const float* __restrict__ b2,
                            float* __restrict__ hid2) {
  int idx = blockIdx.x * blockDim.x + threadIdx.x;
  if (idx >= G_GRAPHS * H2_DIM) return;
  int g = idx / H2_DIM, j = idx % H2_DIM;
  float acc = b2[j];
  const float* hr = hid1 + g * H1_DIM;
  #pragma unroll 4
  for (int k = 0; k < H1_DIM; ++k) acc += hr[k] * W2[k * H2_DIM + j];
  hid2[idx] = acc;                    // no relu (reference)
}

__global__ void mlp3_kernel(const float* __restrict__ hid2,
                            const float* __restrict__ Wo, const float* __restrict__ bo,
                            float* __restrict__ out) {
  int g = blockIdx.x * blockDim.x + threadIdx.x;
  if (g >= G_GRAPHS) return;
  float acc = bo[0];
  const float* hr = hid2 + g * H2_DIM;
  #pragma unroll
  for (int k = 0; k < H2_DIM; ++k) acc += hr[k] * Wo[k];
  out[g] = acc;
}

// ---------------------------------------------------------------------------
// Host-side orchestration (graph-capture safe: only kernel launches)
// ---------------------------------------------------------------------------
extern "C" void kernel_launch(void* const* d_in, const int* in_sizes, int n_in,
                              void* d_out, int out_size, void* d_ws, size_t ws_size,
                              hipStream_t stream) {
  const float* x   = (const float*)d_in[0];
  const int*   ei  = (const int*)  d_in[1];
  const int*   bat = (const int*)  d_in[2];
  const float* Wq  = (const float*)d_in[3];  const float* bq  = (const float*)d_in[4];
  const float* Wk  = (const float*)d_in[5];  const float* bk  = (const float*)d_in[6];
  const float* Wv  = (const float*)d_in[7];  const float* bv  = (const float*)d_in[8];
  const float* Wc3 = (const float*)d_in[9];  const float* bc3 = (const float*)d_in[10];
  const float* Wc5 = (const float*)d_in[11]; const float* bc5 = (const float*)d_in[12];
  const float* Wlt = (const float*)d_in[13]; const float* blt = (const float*)d_in[14];
  const float* Wl  = (const float*)d_in[15]; const float* bl  = (const float*)d_in[16];
  const float* Wr  = (const float*)d_in[17];
  const float* W1  = (const float*)d_in[18]; const float* b1  = (const float*)d_in[19];
  const float* W2  = (const float*)d_in[20]; const float* b2  = (const float*)d_in[21];
  const float* Wo  = (const float*)d_in[22]; const float* bo  = (const float*)d_in[23];
  float* out = (float*)d_out;

  // -------- workspace carve-up --------
  char* ws = (char*)d_ws;
  size_t off = 0;
  auto carve = [&](size_t bytes) -> char* {
    char* p = ws + off;
    off = (off + bytes + 255) & ~(size_t)255;
    return p;
  };
  half_t* Qh  = (half_t*)carve((size_t)NPAD * FPAD * sizeof(half_t));
  half_t* Knh = (half_t*)carve((size_t)NPAD * FPAD * sizeof(half_t));
  half_t* Vt  = (half_t*)carve((size_t)FPAD * NPAD * sizeof(half_t));
  float*  Kf  = (float*) carve((size_t)N_NODES * F_DIM * sizeof(float));
  float*  Kc3 = (float*) carve((size_t)N_NODES * F_DIM * sizeof(float));
  float*  Kc5 = (float*) carve((size_t)N_NODES * F_DIM * sizeof(float));
  float*  h   = (float*) carve((size_t)N_NODES * F_DIM * sizeof(float));
  // zeroed accumulator block (contiguous)
  const int NZ = N_NODES * F_DIM + N_NODES + G_GRAPHS * F_DIM * 2 + G_GRAPHS;
  float* zblock = (float*)carve((size_t)NZ * sizeof(float));
  float* agg   = zblock;
  float* deg   = agg + N_NODES * F_DIM;
  int*   gmaxi = (int*)(deg + N_NODES);
  float* gsum  = (float*)(gmaxi + G_GRAPHS * F_DIM);
  float* gcnt  = gsum + G_GRAPHS * F_DIM;
  float* h2   = (float*)carve((size_t)N_NODES * F_DIM * sizeof(float));
  float* gvec = (float*)carve((size_t)G_GRAPHS * 2 * F_DIM * sizeof(float));
  float* hid1 = (float*)carve((size_t)G_GRAPHS * H1_DIM * sizeof(float));
  float* hid2 = (float*)carve((size_t)G_GRAPHS * H2_DIM * sizeof(float));

  const int B = 256;
  // 1. zero atomic accumulators
  zero_f32_kernel<<<(NZ + B - 1) / B, B, 0, stream>>>(zblock, NZ);
  // 2. projections
  qkv_kernel<<<(NPAD * FPAD + B - 1) / B, B, 0, stream>>>(
      x, Wq, bq, Wk, bk, Wv, bv, Qh, Kf, Vt);
  // 3. conv taps
  conv_kernel<<<(N_NODES * F_DIM + B - 1) / B, B, 0, stream>>>(
      Kf, Wc3, bc3, Wc5, bc5, Kc3, Kc5);
  // 4. K_new
  knew_kernel<<<(NPAD * FPAD + B - 1) / B, B, 0, stream>>>(
      Kc3, Kc5, Kf, Wlt, blt, Knh);
  // 5. fused dense attention (WMMA) -> h
  attn_kernel<<<N_NODES / 16, 32, 0, stream>>>(Qh, Knh, Vt, h);
  // 6. SAGE aggregation
  sage_agg_kernel<<<(E_EDGES + B - 1) / B, B, 0, stream>>>(ei, h, agg, deg);
  // 7. SAGE update
  sage_update_kernel<<<(N_NODES * F_DIM + B - 1) / B, B, 0, stream>>>(
      agg, deg, h, Wl, bl, Wr, h2);
  // 8. pooling
  pool_kernel<<<(N_NODES * F_DIM + B - 1) / B, B, 0, stream>>>(
      bat, h2, gmaxi, gsum, gcnt);
  gvec_kernel<<<(G_GRAPHS * 2 * F_DIM + B - 1) / B, B, 0, stream>>>(
      gmaxi, gsum, gcnt, gvec);
  // 9. head MLP
  mlp1_kernel<<<(G_GRAPHS * H1_DIM + B - 1) / B, B, 0, stream>>>(gvec, W1, b1, hid1);
  mlp2_kernel<<<(G_GRAPHS * H2_DIM + B - 1) / B, B, 0, stream>>>(hid1, W2, b2, hid2);
  mlp3_kernel<<<(G_GRAPHS + B - 1) / B, B, 0, stream>>>(hid2, Wo, bo, out);
}
